// GAT_3229815407223
// MI455X (gfx1250) — compile-verified, matrix operand order checked by reading, not run on hardware
//
#include <hip/hip_runtime.h>
#include <hip/hip_bf16.h>
#include <cstdint>
#include <cstddef>

// ---------------------------------------------------------------------------
// GATv2 x2 for MI455X (gfx1250, wave32).
//  - Dense transforms: bf16 operands pre-packed into WMMA fragment order so
//    the GEMM inner loop is pure {global_load_b128 x2 per operand-frag,
//    v_wmma_f32_16x16x32_bf16}.  Each wave computes a 64x16 tile (4 WMMA / kt).
//  - Sparse softmax-aggregate: coalesced float4 gathers + f32 global atomics;
//    all scatter targets (<60MB) are resident in the 192MB L2.
// ---------------------------------------------------------------------------

typedef __attribute__((ext_vector_type(16))) __bf16 v16bf;
typedef __attribute__((ext_vector_type(8)))  float  v8f;

#define NEG_SLOPE 0.2f

// --------------------------- fragment packing ------------------------------
// Fragment = 512 bf16 covering one 16(row/col) x 32(K) tile, stored so that
// lane l of the consuming wave loads its 16 elements contiguously at
// frag_base + l*16.  Element mapping (CDNA5 ISA 7.12.2, 16-bit A layout):
//   lane l, elem i  ->  r = l & 15,
//                       k = ((l>>4)<<3) + (i & 7) + ((i>>3)<<4)

// Row-major A [R,K] f32 -> A-fragments, frag index (rt * numKt + kt).
// Optional fused ELU (used for the layer-2 input).
__global__ void pack_a_k(const float* __restrict__ in, __bf16* __restrict__ out,
                         int R, int K, int elu) {
  size_t tid = (size_t)blockIdx.x * blockDim.x + threadIdx.x;
  const int numKt = K >> 5;
  const size_t total = (size_t)((R + 15) >> 4) * numKt * 512;
  if (tid >= total) return;
  const int    idx  = (int)(tid & 511);
  const size_t frag = tid >> 9;
  const int kt = (int)(frag % numKt);
  const int rt = (int)(frag / numKt);
  const int l = idx >> 4, i = idx & 15;
  const int row = rt * 16 + (l & 15);
  const int kk  = kt * 32 + ((l >> 4) << 3) + (i & 7) + ((i >> 3) << 4);
  float v = (row < R) ? in[(size_t)row * K + kk] : 0.0f;
  if (elu) v = (v > 0.0f) ? v : (__expf(v) - 1.0f);   // ELU, alpha = 1
  out[tid] = (__bf16)v;
}

// Row-major B [K,M] f32 -> B-fragments, frag index (nt * numKt + kt).
__global__ void pack_b_k(const float* __restrict__ in, __bf16* __restrict__ out,
                         int K, int M) {
  size_t tid = (size_t)blockIdx.x * blockDim.x + threadIdx.x;
  const int numKt = K >> 5;
  const size_t total = (size_t)(M >> 4) * numKt * 512;
  if (tid >= total) return;
  const int    idx  = (int)(tid & 511);
  const size_t frag = tid >> 9;
  const int kt = (int)(frag % numKt);
  const int nt = (int)(frag / numKt);
  const int l = idx >> 4, i = idx & 15;
  const int n  = nt * 16 + (l & 15);
  const int kk = kt * 32 + ((l >> 4) << 3) + (i & 7) + ((i >> 3) << 4);
  out[tid] = (__bf16)in[(size_t)kk * M + n];
}

// ------------------------- WMMA bf16 GEMM (+bias) --------------------------
// C[N,M] = A * B + bias, operands in fragment order.
// Block = 128 threads = 4 waves.  Wave w owns col-tile nt = blockIdx.y*4 + w
// and row-tiles rtBase..rtBase+3 (a 64x16 output tile): per k-step it loads
// 1 B-fragment + 4 A-fragments (contiguous 32B per lane) and issues 4 WMMAs.
__global__ __launch_bounds__(128)
void gemm_frag_wmma_k(const __bf16* __restrict__ Af,
                      const __bf16* __restrict__ Bf,
                      const float*  __restrict__ bias,
                      float* __restrict__ C,
                      int N, int K, int M) {
  const int wave  = threadIdx.x >> 5;
  const int lane  = threadIdx.x & 31;
  const int numKt = K >> 5;
  const int numRt = (N + 15) >> 4;
  const int nt    = blockIdx.y * 4 + wave;
  if (nt * 16 >= M) return;                       // wave-uniform
  const int rtBase = blockIdx.x * 4;

  int rts[4];
#pragma unroll
  for (int t = 0; t < 4; ++t) rts[t] = min(rtBase + t, numRt - 1);

  v8f acc[4] = {v8f{}, v8f{}, v8f{}, v8f{}};
  for (int kt = 0; kt < numKt; ++kt) {
    const v16bf b = *(const v16bf*)(Bf + (((size_t)nt * numKt + kt) << 9) + lane * 16);
#pragma unroll
    for (int t = 0; t < 4; ++t) {
      const v16bf a =
          *(const v16bf*)(Af + (((size_t)rts[t] * numKt + kt) << 9) + lane * 16);
      acc[t] = __builtin_amdgcn_wmma_f32_16x16x32_bf16(
          /*neg_a=*/false, a, /*neg_b=*/false, b,
          /*c_mod=*/(short)0, acc[t], /*reuse_a=*/false, /*reuse_b=*/false);
    }
  }

  // C layout: VGPR r, lane l -> row = rt*16 + r + 8*(l>>4), col = nt*16 + (l&15)
  const int ncol = nt * 16 + (lane & 15);
  const float bv = bias[ncol];
#pragma unroll
  for (int t = 0; t < 4; ++t) {
    const int rbase = (rtBase + t) * 16 + ((lane >> 4) << 3);
#pragma unroll
    for (int r = 0; r < 8; ++r) {
      const int rr = rbase + r;
      if (rr < N) C[(size_t)rr * M + ncol] = acc[t][r] + bv;
    }
  }
}

// ------------------------------ small utils -------------------------------

__global__ void fill_f32_k(float* __restrict__ p, float v, size_t n) {
  size_t i = (size_t)blockIdx.x * blockDim.x + threadIdx.x;
  if (i < n) p[i] = v;
}

// out[i*M + j] = bias[j]   (accumulate buffer starts at bias)
__global__ void init_bias_k(float* __restrict__ out,
                            const float* __restrict__ bias, int N, int M) {
  size_t i = (size_t)blockIdx.x * blockDim.x + threadIdx.x;
  size_t total = (size_t)N * M;
  if (i < total) out[i] = bias[i % M];
}

// ----------------------------- edge kernels --------------------------------

__device__ __forceinline__ void edge_endpoints(const int* __restrict__ ei,
                                               long long E, long long e,
                                               int& s, int& d) {
  if (e < E) { s = ei[e]; d = ei[E + e]; }
  else       { s = d = (int)(e - E); }            // self loops appended
}

__device__ __forceinline__ void atomicMaxF(float* addr, float val) {
  if (val >= 0.0f) atomicMax((int*)addr, __float_as_int(val));
  else             atomicMin((unsigned int*)addr, __float_as_uint(val));
}

// score[e,h] = sum_c leakyrelu(xl[src][h,c] + xr[dst][h,c]) * att[h,c]
// and m[dst,h] = segment-max(score)
__global__ void edge_score_k(const float* __restrict__ xl,
                             const float* __restrict__ xr,
                             const float* __restrict__ att,
                             const int*   __restrict__ ei,
                             long long E, long long E2, int H, int C,
                             float* __restrict__ score,
                             float* __restrict__ mmax) {
  long long t = (long long)blockIdx.x * blockDim.x + threadIdx.x;
  if (t >= E2 * H) return;
  long long e = t / H;
  int h = (int)(t - e * H);
  int s, d;
  edge_endpoints(ei, E, e, s, d);

  const float* pl = xl + (size_t)s * H * C + (size_t)h * C;
  const float* pr = xr + (size_t)d * H * C + (size_t)h * C;
  const float* pa = att + (size_t)h * C;

  float acc = 0.0f;
  for (int c = 0; c < C; c += 4) {
    float4 l = *(const float4*)(pl + c);
    float4 r = *(const float4*)(pr + c);
    float4 a = *(const float4*)(pa + c);
    float v0 = l.x + r.x, v1 = l.y + r.y, v2 = l.z + r.z, v3 = l.w + r.w;
    v0 = (v0 > 0.0f) ? v0 : NEG_SLOPE * v0;
    v1 = (v1 > 0.0f) ? v1 : NEG_SLOPE * v1;
    v2 = (v2 > 0.0f) ? v2 : NEG_SLOPE * v2;
    v3 = (v3 > 0.0f) ? v3 : NEG_SLOPE * v3;
    acc += v0 * a.x + v1 * a.y + v2 * a.z + v3 * a.w;
  }
  score[t] = acc;
  atomicMaxF(&mmax[(size_t)d * H + h], acc);
}

// score[e,h] <- exp(score - m[dst,h]);  denom[dst,h] += score
__global__ void edge_exp_k(const int* __restrict__ ei,
                           long long E, long long E2, int H,
                           float* __restrict__ score,
                           const float* __restrict__ mmax,
                           float* __restrict__ denom) {
  long long t = (long long)blockIdx.x * blockDim.x + threadIdx.x;
  if (t >= E2 * H) return;
  long long e = t / H;
  int h = (int)(t - e * H);
  int s, d;
  edge_endpoints(ei, E, e, s, d);
  float a = __expf(score[t] - mmax[(size_t)d * H + h]);
  score[t] = a;
  atomicAdd(&denom[(size_t)d * H + h], a);
}

// out[dst][h,:] += xl[src][h,:] * (score[e,h] / denom[dst,h])
__global__ void edge_aggregate_k(const float* __restrict__ xl,
                                 const int*   __restrict__ ei,
                                 long long E, long long E2, int H, int C,
                                 const float* __restrict__ score,
                                 const float* __restrict__ denom,
                                 float* __restrict__ out) {
  long long t = (long long)blockIdx.x * blockDim.x + threadIdx.x;
  if (t >= E2 * H) return;
  long long e = t / H;
  int h = (int)(t - e * H);
  int s, d;
  edge_endpoints(ei, E, e, s, d);
  float alpha = score[t] / denom[(size_t)d * H + h];
  const float* pl = xl + (size_t)s * H * C + (size_t)h * C;
  float* po = out + (size_t)d * H * C + (size_t)h * C;
  for (int c = 0; c < C; c += 4) {
    float4 l = *(const float4*)(pl + c);
    atomicAdd(&po[c + 0], l.x * alpha);
    atomicAdd(&po[c + 1], l.y * alpha);
    atomicAdd(&po[c + 2], l.z * alpha);
    atomicAdd(&po[c + 3], l.w * alpha);
  }
}

// ------------------------------ host driver --------------------------------

static inline size_t align256(size_t x) { return (x + 255) & ~(size_t)255; }

extern "C" void kernel_launch(void* const* d_in, const int* in_sizes, int n_in,
                              void* d_out, int out_size, void* d_ws, size_t ws_size,
                              hipStream_t stream) {
  const float* x     = (const float*)d_in[0];
  const int*   ei    = (const int*)  d_in[1];
  const float* Wl1   = (const float*)d_in[2];
  const float* bl1   = (const float*)d_in[3];
  const float* Wr1   = (const float*)d_in[4];
  const float* br1   = (const float*)d_in[5];
  const float* att1  = (const float*)d_in[6];
  const float* bias1 = (const float*)d_in[7];
  const float* Wl2   = (const float*)d_in[8];
  const float* bl2   = (const float*)d_in[9];
  const float* Wr2   = (const float*)d_in[10];
  const float* br2   = (const float*)d_in[11];
  const float* att2  = (const float*)d_in[12];
  const float* bias2 = (const float*)d_in[13];
  float* out = (float*)d_out;

  const int Fin = 128, H1 = 8, C1 = 16, F1 = H1 * C1 /*128*/, C2 = 64;
  const int       N  = in_sizes[0] / Fin;
  const long long E  = in_sizes[1] / 2;
  const long long E2 = E + N;

  const int numRt = (N + 15) >> 4;
  const size_t aFragElems = (size_t)numRt * (Fin >> 5) * 512;  // = N*128 here

  // ---- workspace layout ----
  char* w = (char*)d_ws;
  size_t off = 0;
  __bf16* xbf   = (__bf16*)(w + off); off = align256(off + aFragElems * sizeof(__bf16));
  __bf16* wbf   = (__bf16*)(w + off); off = align256(off + (size_t)Fin * F1 * sizeof(__bf16));
  float*  xl    = (float*) (w + off); off = align256(off + (size_t)N * F1 * sizeof(float));
  float*  xr    = (float*) (w + off); off = align256(off + (size_t)N * F1 * sizeof(float));
  float*  score = (float*) (w + off); off = align256(off + (size_t)E2 * H1 * sizeof(float));
  float*  mbuf  = (float*) (w + off); off = align256(off + (size_t)N * H1 * sizeof(float));
  float*  denom = (float*) (w + off); off = align256(off + (size_t)N * H1 * sizeof(float));
  float*  agg   = (float*) (w + off); off = align256(off + (size_t)N * F1 * sizeof(float));
  (void)ws_size; (void)n_in; (void)out_size;

  const int TB = 256;
  auto blocks = [&](long long n) { return (unsigned)((n + TB - 1) / TB); };
  const dim3 gemm_blk(128, 1, 1);
  const unsigned rowB = (unsigned)((numRt + 3) / 4);   // 64 rows per block

  // ======================= Layer 1 (H=8, C=16) =======================
  pack_a_k<<<blocks((long long)aFragElems), TB, 0, stream>>>(x, xbf, N, Fin, 0);

  pack_b_k<<<blocks(Fin * F1), TB, 0, stream>>>(Wl1, wbf, Fin, F1);
  gemm_frag_wmma_k<<<dim3(rowB, F1 / 64), gemm_blk, 0, stream>>>(xbf, wbf, bl1, xl, N, Fin, F1);
  pack_b_k<<<blocks(Fin * F1), TB, 0, stream>>>(Wr1, wbf, Fin, F1);
  gemm_frag_wmma_k<<<dim3(rowB, F1 / 64), gemm_blk, 0, stream>>>(xbf, wbf, br1, xr, N, Fin, F1);

  fill_f32_k<<<blocks((long long)N * H1), TB, 0, stream>>>(mbuf, -__builtin_huge_valf(), (size_t)N * H1);
  fill_f32_k<<<blocks((long long)N * H1), TB, 0, stream>>>(denom, 0.0f, (size_t)N * H1);
  init_bias_k<<<blocks((long long)N * F1), TB, 0, stream>>>(agg, bias1, N, F1);

  edge_score_k<<<blocks(E2 * H1), TB, 0, stream>>>(xl, xr, att1, ei, E, E2, H1, C1, score, mbuf);
  edge_exp_k<<<blocks(E2 * H1), TB, 0, stream>>>(ei, E, E2, H1, score, mbuf, denom);
  edge_aggregate_k<<<blocks(E2 * H1), TB, 0, stream>>>(xl, ei, E, E2, H1, C1, score, denom, agg);

  // ======================= Layer 2 (H=1, C=64) =======================
  // Re-pack layer-1 output with fused ELU as the next A operand (reuse xbf).
  pack_a_k<<<blocks((long long)aFragElems), TB, 0, stream>>>(agg, xbf, N, F1, 1);

  pack_b_k<<<blocks(F1 * C2), TB, 0, stream>>>(Wl2, wbf, F1, C2);
  gemm_frag_wmma_k<<<dim3(rowB, 1), gemm_blk, 0, stream>>>(xbf, wbf, bl2, xl, N, F1, C2);
  pack_b_k<<<blocks(F1 * C2), TB, 0, stream>>>(Wr2, wbf, F1, C2);
  gemm_frag_wmma_k<<<dim3(rowB, 1), gemm_blk, 0, stream>>>(xbf, wbf, br2, xr, N, F1, C2);

  fill_f32_k<<<blocks((long long)N), TB, 0, stream>>>(mbuf, -__builtin_huge_valf(), (size_t)N);
  fill_f32_k<<<blocks((long long)N), TB, 0, stream>>>(denom, 0.0f, (size_t)N);
  init_bias_k<<<blocks((long long)N * C2), TB, 0, stream>>>(out, bias2, N, C2);

  edge_score_k<<<blocks(E2), TB, 0, stream>>>(xl, xr, att2, ei, E, E2, 1, C2, score, mbuf);
  edge_exp_k<<<blocks(E2), TB, 0, stream>>>(ei, E, E2, 1, score, mbuf, denom);
  edge_aggregate_k<<<blocks(E2), TB, 0, stream>>>(xl, ei, E, E2, 1, C2, score, denom, out);
}
